// ContinuousSheafTransport_71640054497903
// MI455X (gfx1250) — matrix operand order repaired and checked by reading, not compiled, not run on hardware
//
#include <hip/hip_runtime.h>
#include <hip/hip_bf16.h>
#include <math.h>

// Problem constants (match reference)
#define DIM    64
#define EDIM   16
#define NNODES 50000
#define NEDGES 400000
#define NSTEPS 2          // T
#define EULER  2          // 1/STEP

typedef _Float16 half_t;
typedef __attribute__((ext_vector_type(16))) _Float16 v16h;
typedef __attribute__((ext_vector_type(8)))  _Float16 v8h;
typedef __attribute__((ext_vector_type(8)))  float    v8f;

__device__ __forceinline__ v8f wmma16(v16h a, v16h b, v8f c) {
  // D = A(16x32 f16) * B(32x16 f16) + C(16x16 f32)
  return __builtin_amdgcn_wmma_f32_16x16x32_f16(false, a, false, b, (short)0, c,
                                                false, false);
}

// A-fragment (16x32, f16) from LDS staged row-major [16][pitch].
// Lane layout: lanes 0-15 -> M=lane, elems 0..15 = K chunk*32 + 0..15
//              lanes 16-31 -> M=lane-16, K chunk*32 + 16..31.
__device__ __forceinline__ v16h load_a_frag(const half_t* base, int pitch, int c, int lane) {
  int m  = lane & 15;
  int hf = (lane >> 4) & 1;
  const half_t* p = base + m * pitch + c * 32 + hf * 16;
  union { v16h v; v8h h[2]; } u;
  u.h[0] = *(const v8h*)p;        // 16B aligned (pitch multiple of 8 f16)
  u.h[1] = *(const v8h*)(p + 8);
  return u.v;
}

// ---------------------------------------------------------------------------
// Edge message kernel:
//   msg = relu([h[row] | h[col] | ea] @ W1 + b1) @ W2 + b2 ;  agg[row] += msg
// ---------------------------------------------------------------------------
__global__ __launch_bounds__(128) void edge_msg_kernel(
    const float* __restrict__ h,  const float* __restrict__ ea,
    const int*   __restrict__ row, const int* __restrict__ col,
    const float* __restrict__ W1, const float* __restrict__ b1,
    const float* __restrict__ W2, const float* __restrict__ b2,
    float* __restrict__ agg, int nwaves) {
  // Weight fragments, pre-swizzled: frag f (= chunk*4 + ntile), lane l -> 16 contig f16
  __shared__ half_t sW1[20 * 512];      // K padded 144->160 : 5 chunks x 4 ntiles
  __shared__ half_t sW2[8 * 512];       // K=64 : 2 chunks x 4 ntiles
  __shared__ half_t sA[4][16 * 168];    // per-wave A staging, K padded to 160 (pitch 168)
  __shared__ half_t sH[4][16 * 72];     // per-wave hidden staging (pitch 72)

  const int tid  = threadIdx.x;
  const int lane = tid & 31;
  const int wid  = tid >> 5;

  for (int idx = tid; idx < 20 * 512; idx += 128) {
    int f = idx >> 9, rem = idx & 511, l = rem >> 4, i = rem & 15;
    int c = f >> 2, nt = f & 3;
    int k = c * 32 + (l >> 4) * 16 + i;       // B frag: K = chunk*32 + half*16 + i
    int n = nt * 16 + (l & 15);
    sW1[idx] = (half_t)((k < 2 * DIM + EDIM) ? W1[k * DIM + n] : 0.f);
  }
  for (int idx = tid; idx < 8 * 512; idx += 128) {
    int f = idx >> 9, rem = idx & 511, l = rem >> 4, i = rem & 15;
    int c = f >> 2, nt = f & 3;
    int k = c * 32 + (l >> 4) * 16 + i;
    int n = nt * 16 + (l & 15);
    sW2[idx] = (half_t)W2[k * DIM + n];
  }
  __syncthreads();

  const int nlo = lane & 15, hf = lane >> 4;
  half_t* A = sA[wid];
  half_t* H = sH[wid];
  const int gw = blockIdx.x * 4 + wid;
  const int ntiles = NEDGES / 16;

  for (int tile = gw; tile < ntiles; tile += nwaves) {
    const int eb = tile * 16;
    // Stage gathered operands as f16 (coalesced per-row 256B reads)
    for (int idx = lane; idx < 16 * DIM; idx += 32) {
      int m = idx >> 6, c0 = idx & 63;
      A[m * 168 + c0] = (half_t)h[(size_t)row[eb + m] * DIM + c0];
    }
    for (int idx = lane; idx < 16 * DIM; idx += 32) {
      int m = idx >> 6, c0 = idx & 63;
      A[m * 168 + DIM + c0] = (half_t)h[(size_t)col[eb + m] * DIM + c0];
    }
    for (int idx = lane; idx < 16 * 32; idx += 32) {
      int m = idx >> 5, c0 = idx & 31;
      A[m * 168 + 2 * DIM + c0] =
          (half_t)((c0 < EDIM) ? ea[(size_t)(eb + m) * EDIM + c0] : 0.f);
    }
    asm volatile("s_wait_dscnt 0" ::: "memory");   // wave-local LDS RAW fence

    // Layer 1: hidden[16x64] = A[16x160] @ W1[160x64]
    v8f acc[4];
    for (int nt = 0; nt < 4; ++nt)
      for (int r = 0; r < 8; ++r) acc[nt][r] = 0.f;
    for (int c = 0; c < 5; ++c) {
      v16h a = load_a_frag(A, 168, c, lane);
      for (int nt = 0; nt < 4; ++nt) {
        v16h b = *(const v16h*)&sW1[((c * 4 + nt) * 32 + lane) * 16];
        acc[nt] = wmma16(a, b, acc[nt]);
      }
    }
    // bias + ReLU, restage as f16 A-operand
    for (int nt = 0; nt < 4; ++nt) {
      float bias = b1[nt * 16 + nlo];
      for (int r = 0; r < 8; ++r) {
        float v = acc[nt][r] + bias;
        v = v > 0.f ? v : 0.f;
        H[(r + 8 * hf) * 72 + nt * 16 + nlo] = (half_t)v;  // C layout: M = r + 8*half
      }
    }
    asm volatile("s_wait_dscnt 0" ::: "memory");

    // Layer 2: msg[16x64] = hidden[16x64] @ W2[64x64]
    v8f macc[4];
    for (int nt = 0; nt < 4; ++nt)
      for (int r = 0; r < 8; ++r) macc[nt][r] = 0.f;
    for (int c = 0; c < 2; ++c) {
      v16h a = load_a_frag(H, 72, c, lane);
      for (int nt = 0; nt < 4; ++nt) {
        v16h b = *(const v16h*)&sW2[((c * 4 + nt) * 32 + lane) * 16];
        macc[nt] = wmma16(a, b, macc[nt]);
      }
    }
    // bias + scatter-add into agg
    for (int nt = 0; nt < 4; ++nt) {
      float bias = b2[nt * 16 + nlo];
      int n = nt * 16 + nlo;
      for (int r = 0; r < 8; ++r) {
        int dst = row[eb + r + 8 * hf];
        atomicAdd(&agg[(size_t)dst * DIM + n], macc[nt][r] + bias);
      }
    }
  }
}

// ---------------------------------------------------------------------------
// Node update kernel:
//   x = [h | agg] @ Wd1 + bd1 ; LayerNorm ; SiLU ; @ Wd2 + bd2 ; h += 0.5*tanh
//   also zeroes agg for the next Euler step.
// ---------------------------------------------------------------------------
__global__ __launch_bounds__(128) void node_update_kernel(
    float* __restrict__ h, float* __restrict__ agg,
    const float* __restrict__ Wd1, const float* __restrict__ bd1,
    const float* __restrict__ ln_g, const float* __restrict__ ln_b,
    const float* __restrict__ Wd2, const float* __restrict__ bd2,
    int nwaves) {
  __shared__ half_t sWd1[16 * 512];     // K=128 : 4 chunks x 4 ntiles
  __shared__ half_t sWd2[8 * 512];      // K=64  : 2 chunks x 4 ntiles
  __shared__ half_t sA[4][16 * 136];    // per-wave [h|agg] staging (pitch 136)
  __shared__ half_t sH[4][16 * 72];

  const int tid  = threadIdx.x;
  const int lane = tid & 31;
  const int wid  = tid >> 5;

  for (int idx = tid; idx < 16 * 512; idx += 128) {
    int f = idx >> 9, rem = idx & 511, l = rem >> 4, i = rem & 15;
    int c = f >> 2, nt = f & 3;
    int k = c * 32 + (l >> 4) * 16 + i;
    int n = nt * 16 + (l & 15);
    sWd1[idx] = (half_t)Wd1[k * DIM + n];
  }
  for (int idx = tid; idx < 8 * 512; idx += 128) {
    int f = idx >> 9, rem = idx & 511, l = rem >> 4, i = rem & 15;
    int c = f >> 2, nt = f & 3;
    int k = c * 32 + (l >> 4) * 16 + i;
    int n = nt * 16 + (l & 15);
    sWd2[idx] = (half_t)Wd2[k * DIM + n];
  }
  __syncthreads();

  const int nlo = lane & 15, hf = lane >> 4;
  half_t* A = sA[wid];
  half_t* H = sH[wid];
  const int gw = blockIdx.x * 4 + wid;
  const int ntiles = NNODES / 16;      // 3125, exact

  for (int tile = gw; tile < ntiles; tile += nwaves) {
    const int nb = tile * 16;
    for (int idx = lane; idx < 16 * DIM; idx += 32) {
      int m = idx >> 6, c0 = idx & 63;
      A[m * 136 + c0] = (half_t)h[(size_t)(nb + m) * DIM + c0];
    }
    for (int idx = lane; idx < 16 * DIM; idx += 32) {
      int m = idx >> 6, c0 = idx & 63;
      A[m * 136 + DIM + c0] = (half_t)agg[(size_t)(nb + m) * DIM + c0];
    }
    asm volatile("s_wait_dscnt 0" ::: "memory");

    v8f acc[4];
    for (int nt = 0; nt < 4; ++nt)
      for (int r = 0; r < 8; ++r) acc[nt][r] = 0.f;
    for (int c = 0; c < 4; ++c) {
      v16h a = load_a_frag(A, 136, c, lane);
      for (int nt = 0; nt < 4; ++nt) {
        v16h b = *(const v16h*)&sWd1[((c * 4 + nt) * 32 + lane) * 16];
        acc[nt] = wmma16(a, b, acc[nt]);
      }
    }
    // bias, then LayerNorm over the 64-wide row (row M = r + 8*half lives on
    // lanes {half*16 .. half*16+15} x 4 ntiles) via xor-butterfly in each half.
    float s[8], q[8];
    for (int r = 0; r < 8; ++r) { s[r] = 0.f; q[r] = 0.f; }
    for (int nt = 0; nt < 4; ++nt) {
      float bias = bd1[nt * 16 + nlo];
      for (int r = 0; r < 8; ++r) {
        float v = acc[nt][r] + bias;
        acc[nt][r] = v;
        s[r] += v;
        q[r] += v * v;
      }
    }
    for (int off = 1; off < 16; off <<= 1) {
      for (int r = 0; r < 8; ++r) {
        s[r] += __shfl_xor(s[r], off, 32);
        q[r] += __shfl_xor(q[r], off, 32);
      }
    }
    for (int nt = 0; nt < 4; ++nt) {
      int n = nt * 16 + nlo;
      float g = ln_g[n], bb = ln_b[n];
      for (int r = 0; r < 8; ++r) {
        float mean = s[r] * (1.f / 64.f);
        float var  = q[r] * (1.f / 64.f) - mean * mean;
        float rstd = rsqrtf(var + 1e-5f);
        float v = (acc[nt][r] - mean) * rstd * g + bb;
        v = v / (1.f + __expf(-v));                      // SiLU
        H[(r + 8 * hf) * 72 + n] = (half_t)v;
      }
    }
    asm volatile("s_wait_dscnt 0" ::: "memory");

    v8f macc[4];
    for (int nt = 0; nt < 4; ++nt)
      for (int r = 0; r < 8; ++r) macc[nt][r] = 0.f;
    for (int c = 0; c < 2; ++c) {
      v16h a = load_a_frag(H, 72, c, lane);
      for (int nt = 0; nt < 4; ++nt) {
        v16h b = *(const v16h*)&sWd2[((c * 4 + nt) * 32 + lane) * 16];
        macc[nt] = wmma16(a, b, macc[nt]);
      }
    }
    for (int nt = 0; nt < 4; ++nt) {
      int n = nt * 16 + nlo;
      float bias = bd2[n];
      for (int r = 0; r < 8; ++r) {
        int node = nb + r + 8 * hf;
        float v = tanhf(macc[nt][r] + bias);
        size_t o = (size_t)node * DIM + n;
        h[o] = h[o] + 0.5f * v;     // Euler step
        agg[o] = 0.f;               // reset for next step
      }
    }
  }
}

// h = (first ? 2*hs : h + hs)
__global__ void hadd_kernel(float* __restrict__ h, const float* __restrict__ hs, int first) {
  int i = blockIdx.x * blockDim.x + threadIdx.x;
  if (i < NNODES * DIM) h[i] = first ? 2.f * hs[i] : h[i] + hs[i];
}

__global__ void zero_kernel(float* __restrict__ p, int n) {
  int i = blockIdx.x * blockDim.x + threadIdx.x;
  if (i < n) p[i] = 0.f;
}

// D[e] = min(||h[row]-h[col]||^2, 1e4)
__global__ void disagree_kernel(const float* __restrict__ h,
                                const int* __restrict__ row, const int* __restrict__ col,
                                float* __restrict__ outD) {
  int e = blockIdx.x * blockDim.x + threadIdx.x;
  if (e >= NEDGES) return;
  const float4* a = (const float4*)(h + (size_t)row[e] * DIM);
  const float4* b = (const float4*)(h + (size_t)col[e] * DIM);
  float sum = 0.f;
  for (int i = 0; i < DIM / 4; ++i) {
    float4 x = a[i], y = b[i];
    float dx = x.x - y.x, dy = x.y - y.y, dz = x.z - y.z, dw = x.w - y.w;
    sum += dx * dx + dy * dy + dz * dz + dw * dw;
  }
  outD[e] = fminf(sum, 1e4f);
}

extern "C" void kernel_launch(void* const* d_in, const int* in_sizes, int n_in,
                              void* d_out, int out_size, void* d_ws, size_t ws_size,
                              hipStream_t stream) {
  const float* h_seq     = (const float*)d_in[0];   // [T,N,D]
  const float* edge_attr = (const float*)d_in[1];   // [T,E,ED]
  const float* W1  = (const float*)d_in[2];
  const float* b1  = (const float*)d_in[3];
  const float* W2  = (const float*)d_in[4];
  const float* b2  = (const float*)d_in[5];
  const float* Wd1 = (const float*)d_in[6];
  const float* bd1 = (const float*)d_in[7];
  const float* ln_g = (const float*)d_in[8];
  const float* ln_b = (const float*)d_in[9];
  const float* Wd2 = (const float*)d_in[10];
  const float* bd2 = (const float*)d_in[11];
  const int* edge_index = (const int*)d_in[12];     // [T,2,E]

  float* out = (float*)d_out;                       // [N*D] h_final, then [T*E] D
  float* d_h   = (float*)d_ws;                      // [N,D] state
  float* d_agg = d_h + (size_t)NNODES * DIM;        // [N,D] message accumulator

  const int nd = NNODES * DIM;
  zero_kernel<<<(nd + 255) / 256, 256, 0, stream>>>(d_agg, nd);

  const int EBLK = 1024, NBLK = 512;                // 128-thread (4-wave) blocks
  for (int t = 0; t < NSTEPS; ++t) {
    const int* rowp = edge_index + (size_t)t * 2 * NEDGES;
    const int* colp = rowp + NEDGES;
    const float* eat = edge_attr + (size_t)t * NEDGES * EDIM;

    hadd_kernel<<<(nd + 255) / 256, 256, 0, stream>>>(d_h, h_seq + (size_t)t * nd, t == 0);

    for (int s = 0; s < EULER; ++s) {
      edge_msg_kernel<<<EBLK, 128, 0, stream>>>(d_h, eat, rowp, colp,
                                                W1, b1, W2, b2, d_agg, EBLK * 4);
      node_update_kernel<<<NBLK, 128, 0, stream>>>(d_h, d_agg, Wd1, bd1,
                                                   ln_g, ln_b, Wd2, bd2, NBLK * 4);
    }
    disagree_kernel<<<(NEDGES + 127) / 128, 128, 0, stream>>>(
        d_h, rowp, colp, out + (size_t)NNODES * DIM + (size_t)t * NEDGES);
  }
  hipMemcpyAsync(d_out, d_h, (size_t)NNODES * DIM * sizeof(float),
                 hipMemcpyDeviceToDevice, stream);
}
